// GraphIpaDenoiser_66159676228221
// MI455X (gfx1250) — matrix operand order repaired and checked by reading, not compiled, not run on hardware
//
#include <hip/hip_runtime.h>
#include <cstdint>
#include <cstddef>

#define CDIV(a,b) (((a)+(b)-1)/(b))

typedef __attribute__((ext_vector_type(16))) __bf16 v16bf;
typedef __attribute__((ext_vector_type(8)))  float  v8f;

// ---------------- problem dims ----------------
static constexpr int NN   = 3072;      // B*L
static constexpr int BBATCH = 8;
static constexpr int LSEQ = 384;
static constexpr int DEG  = 60;
static constexpr int EE   = NN * DEG;  // 184320
static constexpr int CS   = 256;
static constexpr int CZ   = 128;
static constexpr int HH   = 8;
static constexpr int CHD  = 32;
static constexpr int PV   = 12;
static constexpr int HT   = 64;
static constexpr int NVN  = 4;
static constexpr int CATW = HH * (CHD + CZ + PV * 4); // 1664

static __device__ __forceinline__ int imin(int a, int b) { return a < b ? a : b; }

__device__ __forceinline__ float dot32(const float* __restrict__ a, const float* __restrict__ b) {
  float s = 0.f;
#pragma unroll
  for (int i = 0; i < 32; ++i) s += a[i] * b[i];
  return s;
}

__device__ __forceinline__ void quat_rot(const float q[4], const float v[3], float out[3]) {
  // t = 2*cross(qv, v); out = v + qw*t + cross(qv, t)
  float tx = 2.f * (q[2] * v[2] - q[3] * v[1]);
  float ty = 2.f * (q[3] * v[0] - q[1] * v[2]);
  float tz = 2.f * (q[1] * v[1] - q[2] * v[0]);
  out[0] = v[0] + q[0] * tx + (q[2] * tz - q[3] * ty);
  out[1] = v[1] + q[0] * ty + (q[3] * tx - q[1] * tz);
  out[2] = v[2] + q[0] * tz + (q[1] * ty - q[2] * tx);
}

// ================= WMMA GEMM =================
// Out[M,Nc] = act( A[M,K] @ W[K,Nc] + bias + res ), bf16 compute / f32 accumulate.
// AMODE==1: A row m is concat[z[m,0:128], ne[src[m],0:64], ne[dst[m],0:64]] (K must be 256).
// 64x64 block tile, 8 wave32 (4x2), BK=32, LDS ping-pong, b128 fragment loads.
template<int AMODE>
__global__ void __launch_bounds__(256)
k_gemm(const float* __restrict__ A, const float* __restrict__ W,
       const float* __restrict__ bias, const float* __restrict__ res,
       float* __restrict__ Out, int M, int K, int Nc, int relu,
       const float* __restrict__ ezt, const float* __restrict__ ene,
       const int* __restrict__ esrc, const int* __restrict__ edst)
{
  // padded to 40 elems/row (80B, 16B aligned, conflict-friendly)
  __shared__ __align__(16) __bf16 As[2][64][40];           // As[p][m][k]
  __shared__ __align__(16) __bf16 Bs[2][64][40];           // Bs[p][n][k]  (transposed)
  const int tid  = threadIdx.x;
  const int lane = tid & 31;
  const int wv   = tid >> 5;
  const int wm   = wv & 3;
  const int wn   = wv >> 2;
  const int m0   = blockIdx.x * 64;
  const int n0   = blockIdx.y * 64;
  const int ml   = lane & 15;
  const int hi   = lane >> 4; // 0 or 1

  // staging coordinates: A -> thread owns 8 consecutive k of one row
  const int ar = tid >> 2;           // 0..63 (m within tile)
  const int ac = (tid & 3) * 8;      // 0,8,16,24 (k within tile)
  // B -> thread owns 8 consecutive n of one k-row (coalesced global read)
  const int br = tid >> 3;           // 0..31 (k within tile)
  const int bc = (tid & 7) * 8;      // 0..56 (n within tile)
  const int agm  = imin(m0 + ar, M - 1);
  const bool am_ok = (m0 + ar) < M;

  auto loadA = [&](float* r, int k0) {
    if (AMODE == 0) {
      const float* base = A + (size_t)agm * K;
      if (k0 + ac + 64 < K) __builtin_prefetch(base + k0 + ac + 64, 0, 1);
#pragma unroll
      for (int j = 0; j < 8; ++j) {
        int gk = k0 + ac + j;
        float v = base[imin(gk, K - 1)];
        r[j] = (am_ok && gk < K) ? v : 0.f;
      }
    } else {
      int gk0 = k0 + ac;               // region boundaries (128/192) are 8-aligned
      const float* base;
      if (gk0 < 128)      base = ezt + (size_t)agm * 128 + gk0;
      else if (gk0 < 192) base = ene + (size_t)esrc[agm] * 64 + (gk0 - 128);
      else                base = ene + (size_t)edst[agm] * 64 + (gk0 - 192);
#pragma unroll
      for (int j = 0; j < 8; ++j) {
        float v = base[j];
        r[j] = am_ok ? v : 0.f;
      }
    }
  };
  auto storeA = [&](const float* r, int p) {
    union { __bf16 h[8]; uint4 q; } pk;
#pragma unroll
    for (int j = 0; j < 8; ++j) pk.h[j] = (__bf16)r[j];
    *(uint4*)&As[p][ar][ac] = pk.q;    // one 16B LDS store
  };
  auto loadB = [&](float* r, int k0) {
    int gk = k0 + br;
    const float* base = W + (size_t)imin(gk, K - 1) * Nc;
#pragma unroll
    for (int j = 0; j < 8; ++j) {
      int gn = n0 + bc + j;
      float v = base[imin(gn, Nc - 1)];
      r[j] = (gk < K && gn < Nc) ? v : 0.f;
    }
  };
  auto storeB = [&](const float* r, int p) {
#pragma unroll
    for (int j = 0; j < 8; ++j) Bs[p][bc + j][br] = (__bf16)r[j];
  };

  v8f acc0 = {0.f,0.f,0.f,0.f,0.f,0.f,0.f,0.f};
  v8f acc1 = {0.f,0.f,0.f,0.f,0.f,0.f,0.f,0.f};

  float ra[8], rb[8];
  loadA(ra, 0); loadB(rb, 0);
  storeA(ra, 0); storeB(rb, 0);
  int p = 0;
  for (int k0 = 0; k0 < K; k0 += 32) {
    __syncthreads();
    const bool more = (k0 + 32) < K;
    if (more) { loadA(ra, k0 + 32); loadB(rb, k0 + 32); }  // overlap with WMMA

    union { v16bf v; uint4 q[2]; } fa, fb0, fb1;
    // A fragment (ISA 16-bit 16x32 layout): two contiguous 16B runs per lane
    fa.q[0]  = *(const uint4*)&As[p][wm * 16 + ml][hi * 8];
    fa.q[1]  = *(const uint4*)&As[p][wm * 16 + ml][16 + hi * 8];
    // B fragment (ISA 16-bit 32x16 layout) from transposed tile: two 16B runs, same k order as A
    fb0.q[0] = *(const uint4*)&Bs[p][wn * 32 + ml][hi * 8];
    fb0.q[1] = *(const uint4*)&Bs[p][wn * 32 + ml][16 + hi * 8];
    fb1.q[0] = *(const uint4*)&Bs[p][wn * 32 + 16 + ml][hi * 8];
    fb1.q[1] = *(const uint4*)&Bs[p][wn * 32 + 16 + ml][16 + hi * 8];

    acc0 = __builtin_amdgcn_wmma_f32_16x16x32_bf16(false, fa.v, false, fb0.v, (short)0, acc0, false, false);
    acc1 = __builtin_amdgcn_wmma_f32_16x16x32_bf16(false, fa.v, false, fb1.v, (short)0, acc1, false, false);

    if (more) { storeA(ra, p ^ 1); storeB(rb, p ^ 1); }
    p ^= 1;
  }

  const int rb0 = m0 + wm * 16 + hi * 8;
  const int c0  = n0 + wn * 32 + ml;
  const int c1  = c0 + 16;
#pragma unroll
  for (int r = 0; r < 8; ++r) {
    int gr = rb0 + r;
    if (gr >= M) continue;
    if (c0 < Nc) {
      float v = acc0[r] + bias[c0];
      if (res)  v += res[(size_t)gr * Nc + c0];
      if (relu) v = fmaxf(v, 0.f);
      Out[(size_t)gr * Nc + c0] = v;
    }
    if (c1 < Nc) {
      float v = acc1[r] + bias[c1];
      if (res)  v += res[(size_t)gr * Nc + c1];
      if (relu) v = fmaxf(v, 0.f);
      Out[(size_t)gr * Nc + c1] = v;
    }
  }
}

// ================= LayerNorm (optional residual), in-place safe =================
__global__ void __launch_bounds__(128)
k_ln(float* __restrict__ out, const float* __restrict__ x, const float* __restrict__ res,
     const float* __restrict__ g, const float* __restrict__ b, int D)
{
  const long row = blockIdx.x;
  const float* xr = x + row * (long)D;
  const float* rr = res ? res + row * (long)D : nullptr;
  float s = 0.f, s2 = 0.f;
  for (int j = threadIdx.x; j < D; j += 128) {
    float v = xr[j] + (rr ? rr[j] : 0.f);
    s += v; s2 += v * v;
  }
  __shared__ float r1[128], r2[128];
  r1[threadIdx.x] = s; r2[threadIdx.x] = s2;
  __syncthreads();
  for (int o = 64; o > 0; o >>= 1) {
    if (threadIdx.x < o) { r1[threadIdx.x] += r1[threadIdx.x + o]; r2[threadIdx.x] += r2[threadIdx.x + o]; }
    __syncthreads();
  }
  float mean = r1[0] / (float)D;
  float var  = r2[0] / (float)D - mean * mean;
  float inv  = rsqrtf(var + 1e-5f);
  float* orow = out + row * (long)D;
  for (int j = threadIdx.x; j < D; j += 128) {
    float v = xr[j] + (rr ? rr[j] : 0.f);
    orow[j] = (v - mean) * inv * g[j] + b[j];
  }
}

// ================= misc elementwise / graph kernels =================
__global__ void k_norm_rows(float* __restrict__ out, const float* __restrict__ in, int rows, int d) {
  int id = blockIdx.x * blockDim.x + threadIdx.x;
  if (id >= rows) return;
  float ss = 0.f;
  for (int j = 0; j < d; ++j) { float v = in[(long)id * d + j]; ss += v * v; }
  float inv = rsqrtf(ss);
  for (int j = 0; j < d; ++j) out[(long)id * d + j] = in[(long)id * d + j] * inv;
}

__global__ void __launch_bounds__(128) k_center(const float* __restrict__ trans, float* __restrict__ center) {
  int b = blockIdx.x;
  float sx = 0.f, sy = 0.f, sz = 0.f;
  for (int n = threadIdx.x; n < LSEQ; n += 128) {
    const float* p = trans + ((long)b * LSEQ + n) * 3;
    sx += p[0]; sy += p[1]; sz += p[2];
  }
  __shared__ float rx[128], ry[128], rz[128];
  rx[threadIdx.x] = sx; ry[threadIdx.x] = sy; rz[threadIdx.x] = sz;
  __syncthreads();
  for (int o = 64; o > 0; o >>= 1) {
    if (threadIdx.x < o) {
      rx[threadIdx.x] += rx[threadIdx.x + o];
      ry[threadIdx.x] += ry[threadIdx.x + o];
      rz[threadIdx.x] += rz[threadIdx.x + o];
    }
    __syncthreads();
  }
  if (threadIdx.x == 0) {
    float inv = 1.f / ((float)LSEQ + 1e-8f);
    center[b * 3 + 0] = rx[0] * inv;
    center[b * 3 + 1] = ry[0] * inv;
    center[b * 3 + 2] = rz[0] * inv;
  }
}

__global__ void k_center_sub(float* __restrict__ tc, const float* __restrict__ tin,
                             const float* __restrict__ center) {
  int id = blockIdx.x * blockDim.x + threadIdx.x;
  if (id >= NN * 3) return;
  int n = id / 3, c = id % 3;
  tc[id] = tin[id] - center[(n / LSEQ) * 3 + c];
}

__global__ void k_scale(float* __restrict__ p, float s, int n) {
  int id = blockIdx.x * blockDim.x + threadIdx.x;
  if (id < n) p[id] *= s;
}

__global__ void k_final_trans(float* __restrict__ out, const float* __restrict__ tc,
                              const float* __restrict__ center) {
  int id = blockIdx.x * blockDim.x + threadIdx.x;
  if (id >= NN * 3) return;
  int n = id / 3, c = id % 3;
  out[id] = tc[id] * 10.f + center[(n / LSEQ) * 3 + c];
}

__global__ void k_edge_feat(const int* __restrict__ src, const int* __restrict__ dst,
                            const float* __restrict__ tc, float* __restrict__ ef) {
  int e = blockIdx.x * blockDim.x + threadIdx.x;
  if (e >= EE) return;
  int s = src[e], d = dst[e];
  float dx = tc[s * 3 + 0] - tc[d * 3 + 0];
  float dy = tc[s * 3 + 1] - tc[d * 3 + 1];
  float dz = tc[s * 3 + 2] - tc[d * 3 + 2];
  float dist = sqrtf(dx * dx + dy * dy + dz * dz);
  float* o = ef + (long)e * CZ;
  const float sig = 20.f / 64.f;
  for (int j = 0; j < 64; ++j) {
    float mu = 20.f * (float)j / 63.f;
    float r = (dist - mu) / sig;
    o[j] = expf(-r * r);
  }
  float x = (float)(s % LSEQ) - (float)(d % LSEQ);
  const float lg = logf(10000.f);
  for (int j = 0; j < 32; ++j) {
    float f = expf(-lg * (float)j / 32.f);
    float a = x * f;
    o[64 + j] = cosf(a);
    o[96 + j] = sinf(a);
  }
}

__global__ void k_sins_t(const float* __restrict__ t, float* __restrict__ ft) {
  int id = blockIdx.x * blockDim.x + threadIdx.x;
  if (id >= BBATCH * HT) return;
  int b = id >> 6, j = id & 63;
  float x = t[b] * 1000.f;
  const float lg = logf(10000.f);
  int jj = (j < 32) ? j : (j - 32);
  float a = x * expf(-lg * (float)jj / 32.f);
  ft[id] = (j < 32) ? cosf(a) : sinf(a);
}

__global__ void k_node_in(const float* __restrict__ ft, float* __restrict__ out) {
  int id = blockIdx.x * blockDim.x + threadIdx.x;
  if (id >= NN * 321) return;
  int n = id / 321, j = id % 321;
  float v;
  if (j < 256) {
    const float lg = logf(10000.f);
    int jj = (j < 128) ? j : (j - 128);
    float a = (float)(n % LSEQ) * expf(-lg * (float)jj / 128.f);
    v = (j < 128) ? cosf(a) : sinf(a);
  } else if (j < 320) {
    v = ft[(n / LSEQ) * HT + (j - 256)];
  } else {
    v = 1.f; // noising mask (all ones)
  }
  out[id] = v;
}

__global__ void k_concat2(float* __restrict__ out, const float* __restrict__ a,
                          const float* __restrict__ b) {
  int id = blockIdx.x * blockDim.x + threadIdx.x;
  if (id >= NN * 384) return;
  int n = id / 384, j = id % 384;
  out[id] = (j < 256) ? a[(long)n * 256 + j] : b[(long)n * 128 + (j - 256)];
}

__global__ void k_to_glob(float* __restrict__ buf, const float* __restrict__ quat,
                          const float* __restrict__ trans, int npts) {
  int id = blockIdx.x * blockDim.x + threadIdx.x;
  if (id >= NN * npts) return;
  int n = id / npts, p = id % npts;
  float* ptr = buf + ((long)n * npts + p) * 3;
  float v[3] = {ptr[0], ptr[1], ptr[2]}, r[3];
  float q[4] = {quat[n * 4], quat[n * 4 + 1], quat[n * 4 + 2], quat[n * 4 + 3]};
  quat_rot(q, v, r);
  ptr[0] = r[0] + trans[n * 3 + 0];
  ptr[1] = r[1] + trans[n * 3 + 1];
  ptr[2] = r[2] + trans[n * 3 + 2];
}

__global__ void k_ipa_logits(const float* __restrict__ q, const float* __restrict__ k,
                             const float* __restrict__ qpg, const float* __restrict__ kpg,
                             const float* __restrict__ bij, const float* __restrict__ hwp,
                             const int* __restrict__ src, const int* __restrict__ dst,
                             float* __restrict__ lg) {
  int id = blockIdx.x * blockDim.x + threadIdx.x;
  if (id >= EE * HH) return;
  int e = id >> 3, h = id & 7;
  int s = src[e], d = dst[e];
  float sc = dot32(q + (long)d * CS + h * 32, k + (long)s * CS + h * 32) * 0.1767766953f;
  float d2 = 0.f;
  const float* qp = qpg + (long)d * 192 + h * 24;
  const float* kp = kpg + (long)s * 192 + h * 24;
#pragma unroll
  for (int i = 0; i < 24; ++i) { float df = qp[i] - kp[i]; d2 += df * df; }
  float hw = log1pf(expf(hwp[h]));
  const float wL = 0.5773502692f;
  lg[id] = wL * (sc + bij[id]) - 0.5f * wL * (1.f / 6.f) * hw * d2;
}

__global__ void k_seg_softmax_deg(const float* __restrict__ lg, float* __restrict__ att) {
  int id = blockIdx.x * blockDim.x + threadIdx.x;
  if (id >= NN * HH) return;
  int n = id >> 3, h = id & 7;
  const float* p = lg + (long)n * (DEG * HH) + h;
  float m = -3e38f;
  for (int j = 0; j < DEG; ++j) m = fmaxf(m, p[j * 8]);
  if (!(m > -3e37f)) m = 0.f;
  float s = 0.f;
  for (int j = 0; j < DEG; ++j) s += expf(p[j * 8] - m);
  float inv = 1.f / (s + 1e-9f);
  float* a = att + (long)n * (DEG * HH) + h;
  for (int j = 0; j < DEG; ++j) a[j * 8] = expf(p[j * 8] - m) * inv;
}

__global__ void k_ipa_o(const float* __restrict__ att, const float* __restrict__ v,
                        const int* __restrict__ src, float* __restrict__ cat) {
  int id = blockIdx.x * blockDim.x + threadIdx.x;
  if (id >= NN * CS) return;
  int n = id >> 8, hc = id & 255;
  const float* ab = att + (long)n * 480 + (hc >> 5);
  float s = 0.f;
  for (int j = 0; j < DEG; ++j) {
    int e = n * DEG + j;
    s += ab[j * 8] * v[(long)src[e] * CS + hc];
  }
  cat[(long)n * CATW + hc] = s;
}

__global__ void k_ipa_opt(const float* __restrict__ att, const float* __restrict__ vpg,
                          const int* __restrict__ src, const float* __restrict__ quat,
                          const float* __restrict__ trans, float* __restrict__ cat) {
  int id = blockIdx.x * blockDim.x + threadIdx.x;
  if (id >= NN * 96) return;
  int n = id / 96, hp = id % 96;
  int h = hp / 12;
  const float* ab = att + (long)n * 480 + h;
  float sx = 0.f, sy = 0.f, sz = 0.f;
  for (int j = 0; j < DEG; ++j) {
    int e = n * DEG + j;
    float a = ab[j * 8];
    const float* vp = vpg + (long)src[e] * 288 + hp * 3;
    sx += a * vp[0]; sy += a * vp[1]; sz += a * vp[2];
  }
  float qi[4] = {quat[n * 4], -quat[n * 4 + 1], -quat[n * 4 + 2], -quat[n * 4 + 3]};
  float vv[3] = {sx - trans[n * 3], sy - trans[n * 3 + 1], sz - trans[n * 3 + 2]};
  float r[3];
  quat_rot(qi, vv, r);
  float* c = cat + (long)n * CATW;
  c[256 + hp * 3 + 0] = r[0];
  c[256 + hp * 3 + 1] = r[1];
  c[256 + hp * 3 + 2] = r[2];
  c[544 + hp] = sqrtf(r[0] * r[0] + r[1] * r[1] + r[2] * r[2] + 1e-8f);
}

__global__ void k_ipa_opair(const float* __restrict__ att, const float* __restrict__ z,
                            float* __restrict__ cat) {
  int id = blockIdx.x * blockDim.x + threadIdx.x;
  if (id >= NN * 1024) return;
  int n = id >> 10, hc = id & 1023;
  int h = hc >> 7, c = hc & 127;
  const float* ab = att + (long)n * 480 + h;
  const float* zb = z + (long)n * DEG * CZ + c;
  float s = 0.f;
  for (int j = 0; j < DEG; ++j) s += ab[j * 8] * zb[j * CZ];
  cat[(long)n * CATW + 640 + hc] = s;
}

__global__ void k_vn_lg(const float* __restrict__ kn, const float* __restrict__ qv,
                        float* __restrict__ lgv) {
  int id = blockIdx.x * blockDim.x + threadIdx.x;
  if (id >= NN * 32) return;
  int n = id >> 5, vh = id & 31;
  int v = vh >> 3, h = vh & 7;
  int b = n / LSEQ;
  lgv[id] = dot32(kn + (long)n * CS + h * 32,
                  qv + (long)(b * NVN + v) * CS + h * 32) * 0.1767766953f;
}

__global__ void __launch_bounds__(128)
k_vn_softmax(const float* __restrict__ lgv, float* __restrict__ avn) {
  int bc = blockIdx.x;
  int b = bc >> 5, col = bc & 31;
  const float* base = lgv + (long)b * LSEQ * 32 + col;
  float m = -3e38f;
  for (int n = threadIdx.x; n < LSEQ; n += 128) m = fmaxf(m, base[(long)n * 32]);
  __shared__ float r[128];
  r[threadIdx.x] = m; __syncthreads();
  for (int o = 64; o > 0; o >>= 1) {
    if (threadIdx.x < o) r[threadIdx.x] = fmaxf(r[threadIdx.x], r[threadIdx.x + o]);
    __syncthreads();
  }
  m = r[0];
  if (!(m > -3e37f)) m = 0.f;
  __syncthreads();
  float s = 0.f;
  for (int n = threadIdx.x; n < LSEQ; n += 128) s += expf(base[(long)n * 32] - m);
  r[threadIdx.x] = s; __syncthreads();
  for (int o = 64; o > 0; o >>= 1) {
    if (threadIdx.x < o) r[threadIdx.x] += r[threadIdx.x + o];
    __syncthreads();
  }
  float d = r[0];
  float* ab = avn + (long)b * LSEQ * 32 + col;
  for (int n = threadIdx.x; n < LSEQ; n += 128)
    ab[(long)n * 32] = expf(base[(long)n * 32] - m) / (d + 1e-9f);
}

__global__ void k_vn_agg(const float* __restrict__ avn, const float* __restrict__ vnv,
                         float* __restrict__ agg) {
  int id = blockIdx.x * blockDim.x + threadIdx.x;
  if (id >= BBATCH * NVN * HH * 32) return;
  int b = id >> 10;
  int rem = id & 1023;
  int v = rem >> 8;
  int hd = rem & 255;           // h*32 + d
  int vh = v * 8 + (hd >> 5);
  float s = 0.f;
  for (int n = 0; n < LSEQ; ++n) {
    long row = (long)b * LSEQ + n;
    s += avn[row * 32 + vh] * vnv[row * CS + hd];
  }
  agg[id] = s;
}

__global__ void k_vn_attn2(const float* __restrict__ qn, const float* __restrict__ kv,
                           const float* __restrict__ vv, float* __restrict__ agg2) {
  int id = blockIdx.x * blockDim.x + threadIdx.x;
  if (id >= NN * HH) return;
  int n = id >> 3, h = id & 7;
  int b = n / LSEQ;
  float l[NVN];
  float m = -3e38f;
#pragma unroll
  for (int v = 0; v < NVN; ++v) {
    l[v] = dot32(qn + (long)n * CS + h * 32,
                 kv + (long)(b * NVN + v) * CS + h * 32) * 0.1767766953f;
    m = fmaxf(m, l[v]);
  }
  float s = 0.f;
#pragma unroll
  for (int v = 0; v < NVN; ++v) { l[v] = expf(l[v] - m); s += l[v]; }
  float inv = 1.f / s;
  for (int d = 0; d < 32; ++d) {
    float acc = 0.f;
#pragma unroll
    for (int v = 0; v < NVN; ++v) acc += l[v] * vv[(long)(b * NVN + v) * CS + h * 32 + d];
    agg2[(long)n * CS + h * 32 + d] = acc * inv;
  }
}

__global__ void k_bb_update(const float* __restrict__ u6, float* __restrict__ quat,
                            float* __restrict__ trans) {
  int id = blockIdx.x * blockDim.x + threadIdx.x;
  if (id >= NN) return;
  const float* u = u6 + (long)id * 6;
  float* qp = quat + id * 4;
  float* tp = trans + id * 3;
  float q0 = qp[0], q1 = qp[1], q2 = qp[2], q3 = qp[3];
  float inv = rsqrtf(1.f + u[0] * u[0] + u[1] * u[1] + u[2] * u[2]);
  float w = inv, x = u[0] * inv, y = u[1] * inv, z = u[2] * inv;
  float vold[3] = {u[3], u[4], u[5]}, r[3];
  float qo[4] = {q0, q1, q2, q3};
  quat_rot(qo, vold, r);            // new_trans uses OLD quats
  tp[0] += r[0]; tp[1] += r[1]; tp[2] += r[2];
  qp[0] = q0 * w - q1 * x - q2 * y - q3 * z;
  qp[1] = q0 * x + q1 * w + q2 * z - q3 * y;
  qp[2] = q0 * y - q1 * z + q2 * w + q3 * x;
  qp[3] = q0 * z + q1 * y - q2 * x + q3 * w;
}

// ================= host orchestration =================
struct Lin { const float* w; const float* b; };
struct LNP { const float* g; const float* b; };

static void gemm(hipStream_t st, const float* A, const Lin& L, float* out,
                 int M, int K, int Nc, int relu, const float* res = nullptr) {
  dim3 g(CDIV(M, 64), CDIV(Nc, 64));
  k_gemm<0><<<g, dim3(256), 0, st>>>(A, L.w, L.b, res, out, M, K, Nc, relu,
                                     nullptr, nullptr, nullptr, nullptr);
}
static void gemm_edgecat(hipStream_t st, const float* z, const float* ne,
                         const int* src, const int* dst, const Lin& L, float* out,
                         int M, int Nc, int relu) {
  dim3 g(CDIV(M, 64), CDIV(Nc, 64));
  k_gemm<1><<<g, dim3(256), 0, st>>>(nullptr, L.w, L.b, nullptr, out, M, 256, Nc, relu,
                                     z, ne, src, dst);
}
static void ln(hipStream_t st, float* out, const float* x, const float* res,
               const LNP& p, int rows, int D) {
  k_ln<<<dim3(rows), dim3(128), 0, st>>>(out, x, res, p.g, p.b, D);
}

extern "C" void kernel_launch(void* const* d_in, const int* in_sizes, int n_in,
                              void* d_out, int out_size, void* d_ws, size_t ws_size,
                              hipStream_t stream) {
  (void)in_sizes; (void)n_in; (void)out_size; (void)ws_size;
  const float* in_quats = (const float*)d_in[0];
  const float* in_trans = (const float*)d_in[1];
  const float* in_sc    = (const float*)d_in[2];
  const float* in_t     = (const float*)d_in[3];
  const int*   e_src    = (const int*)d_in[7];
  const int*   e_dst    = e_src + EE;

  // ---- param binding (setup_inputs / make_params insertion order) ----
  int cur = 8;
  auto F = [&]() -> const float* { return (const float*)d_in[cur++]; };
  auto getLin = [&]() { Lin l; l.w = F(); l.b = F(); return l; };
  auto getLN  = [&]() { LNP l; l.g = F(); l.b = F(); return l; };

  Lin en0 = getLin(), en1 = getLin(), en2 = getLin();  LNP enln = getLN();
  Lin ee0 = getLin(), ee1 = getLin(), ee2 = getLin();  LNP eeln = getLN();
  Lin ev0 = getLin(), ev1 = getLin(), ev2 = getLin();  LNP evln = getLN();
  Lin ta1 = getLin(), ta2 = getLin();

  struct Layer {
    Lin fuse, q, k, v, qp, kp, vp, bij, out;
    const float* head_w;
    LNP ln_s1;
    Lin q_vn, k_n, v_n, o_vn; LNP ln_vn;
    Lin q_n, k_vn, v_vn, o_n; LNP ln_n;
    Lin nt1, nt2, nt3; LNP nt_ln;
    Lin bb, sc, et_node, et1, et2; LNP et_ln;
  } Ls[4];
  for (int l = 0; l < 4; ++l) {
    Layer& L = Ls[l];
    L.fuse = getLin();
    L.q = getLin(); L.k = getLin(); L.v = getLin();
    L.qp = getLin(); L.kp = getLin(); L.vp = getLin();
    L.bij = getLin(); L.head_w = F(); L.out = getLin();
    L.ln_s1 = getLN();
    L.q_vn = getLin(); L.k_n = getLin(); L.v_n = getLin(); L.o_vn = getLin();
    L.ln_vn = getLN();
    L.q_n = getLin(); L.k_vn = getLin(); L.v_vn = getLin(); L.o_n = getLin();
    L.ln_n = getLN();
    L.nt1 = getLin(); L.nt2 = getLin(); L.nt3 = getLin(); L.nt_ln = getLN();
    L.bb = getLin(); L.sc = getLin(); L.et_node = getLin();
    L.et1 = getLin(); L.et2 = getLin(); L.et_ln = getLN();
  }

  // ---- d_out regions ----
  float* o_base  = (float*)d_out;
  float* s_out   = o_base;                         // N*256
  float* q_out   = s_out + (size_t)NN * CS;        // N*4
  float* t_out   = q_out + (size_t)NN * 4;         // N*3
  float* sc_out  = t_out + (size_t)NN * 3;         // N*128
  float* psi_out = sc_out + (size_t)NN * CZ;       // N*2
  float* z_out   = psi_out + (size_t)NN * 2;       // E*128

  // ---- workspace bump allocator ----
  float* wsf = (float*)d_ws;
  size_t off = 0;
  auto alloc = [&](size_t n) { float* p = wsf + off; off += n; return p; };

  float* ebuf1  = alloc((size_t)EE * 256);
  float* ebuf2  = alloc((size_t)EE * 128);
  float* lg     = alloc((size_t)EE * 8);
  float* att    = alloc((size_t)EE * 8);
  float* bij    = alloc((size_t)EE * 8);
  float* trans_c= alloc((size_t)NN * 3);
  float* center = alloc(BBATCH * 3);
  float* ft     = alloc(BBATCH * HT);
  float* nodein = alloc((size_t)NN * 321);
  float* vn     = alloc(32 * 256);
  float* vn_qv  = alloc(32 * 256);
  float* vn_agg = alloc(32 * 256);
  float* vn_tmp = alloc(32 * 1024);   // also used as 8x1024 embed tmp
  float* vn_kv  = alloc(32 * 256);
  float* vn_vv  = alloc(32 * 256);
  float* nbuf1  = alloc((size_t)NN * 256);
  float* nbuf2  = alloc((size_t)NN * 256);
  float* inp    = alloc((size_t)NN * 256);
  float* catss  = alloc((size_t)NN * 384);
  float* qb     = alloc((size_t)NN * 256);
  float* kb     = alloc((size_t)NN * 256);
  float* vb     = alloc((size_t)NN * 256);
  float* qpg    = alloc((size_t)NN * 192);
  float* kpg    = alloc((size_t)NN * 192);
  float* vpg    = alloc((size_t)NN * 288);
  float* cat    = alloc((size_t)NN * CATW);
  float* upd    = alloc((size_t)NN * 256);
  float* kn     = alloc((size_t)NN * 256);
  float* vnv    = alloc((size_t)NN * 256);
  float* qn     = alloc((size_t)NN * 256);
  float* agg2   = alloc((size_t)NN * 256);
  float* lgv    = alloc((size_t)NN * 32);
  float* avn    = alloc((size_t)NN * 32);
  float* u6     = alloc((size_t)NN * 6);
  float* ne     = alloc((size_t)NN * 64);

  const int TB = 256;
  // ---- preamble ----
  k_norm_rows<<<CDIV(NN, TB), TB, 0, stream>>>(q_out, in_quats, NN, 4);
  k_center<<<BBATCH, 128, 0, stream>>>(in_trans, center);
  k_center_sub<<<CDIV(NN * 3, TB), TB, 0, stream>>>(trans_c, in_trans, center);
  k_edge_feat<<<CDIV(EE, 128), 128, 0, stream>>>(e_src, e_dst, trans_c, ebuf2);
  k_scale<<<CDIV(NN * 3, TB), TB, 0, stream>>>(trans_c, 0.1f, NN * 3);
  k_sins_t<<<CDIV(BBATCH * HT, TB), TB, 0, stream>>>(in_t, ft);

  // vn embed: mlp3 over ft (8 x 64)
  gemm(stream, ft, ev0, vn_qv, BBATCH, HT, CS, 1);
  gemm(stream, vn_qv, ev1, vn_agg, BBATCH, CS, CS, 1);
  gemm(stream, vn_agg, ev2, vn_tmp, BBATCH, CS, CS * NVN, 0);
  ln(stream, vn, vn_tmp, nullptr, evln, BBATCH, CS * NVN);

  // node embed
  k_node_in<<<CDIV(NN * 321, TB), TB, 0, stream>>>(ft, nodein);
  gemm(stream, nodein, en0, nbuf1, NN, 321, CS, 1);
  gemm(stream, nbuf1, en1, nbuf2, NN, CS, CS, 1);
  gemm(stream, nbuf2, en2, nbuf1, NN, CS, CS, 0);
  ln(stream, s_out, nbuf1, nullptr, enln, NN, CS);

  // edge embed
  gemm(stream, ebuf2, ee0, ebuf1, EE, CZ, CZ, 1);
  gemm(stream, ebuf1, ee1, ebuf2, EE, CZ, CZ, 1);
  gemm(stream, ebuf2, ee2, z_out, EE, CZ, CZ, 0);
  ln(stream, z_out, z_out, nullptr, eeln, EE, CZ);

  hipMemcpyAsync(sc_out, in_sc, (size_t)NN * CZ * sizeof(float),
                 hipMemcpyDeviceToDevice, stream);

  // ---- layers ----
  for (int l = 0; l < 4; ++l) {
    const Layer& L = Ls[l];
    k_concat2<<<CDIV(NN * 384, TB), TB, 0, stream>>>(catss, s_out, sc_out);
    gemm(stream, catss, L.fuse, inp, NN, 384, CS, 0);

    // IPA
    gemm(stream, inp, L.q, qb, NN, CS, CS, 0);
    gemm(stream, inp, L.k, kb, NN, CS, CS, 0);
    gemm(stream, inp, L.v, vb, NN, CS, CS, 0);
    gemm(stream, inp, L.qp, qpg, NN, CS, 192, 0);
    k_to_glob<<<CDIV(NN * 64, TB), TB, 0, stream>>>(qpg, q_out, trans_c, 64);
    gemm(stream, inp, L.kp, kpg, NN, CS, 192, 0);
    k_to_glob<<<CDIV(NN * 64, TB), TB, 0, stream>>>(kpg, q_out, trans_c, 64);
    gemm(stream, inp, L.vp, vpg, NN, CS, 288, 0);
    k_to_glob<<<CDIV(NN * 96, TB), TB, 0, stream>>>(vpg, q_out, trans_c, 96);
    gemm(stream, z_out, L.bij, bij, EE, CZ, HH, 0);
    k_ipa_logits<<<CDIV(EE * HH, TB), TB, 0, stream>>>(qb, kb, qpg, kpg, bij,
                                                       L.head_w, e_src, e_dst, lg);
    k_seg_softmax_deg<<<CDIV(NN * HH, TB), TB, 0, stream>>>(lg, att);
    k_ipa_o<<<CDIV(NN * CS, TB), TB, 0, stream>>>(att, vb, e_src, cat);
    k_ipa_opt<<<CDIV(NN * 96, TB), TB, 0, stream>>>(att, vpg, e_src, q_out, trans_c, cat);
    k_ipa_opair<<<CDIV(NN * 1024, TB), TB, 0, stream>>>(att, z_out, cat);
    gemm(stream, cat, L.out, upd, NN, CATW, CS, 0);
    ln(stream, s_out, s_out, upd, L.ln_s1, NN, CS);

    // virtual-node update
    gemm(stream, vn, L.q_vn, vn_qv, 32, CS, CS, 0);
    gemm(stream, s_out, L.k_n, kn, NN, CS, CS, 0);
    gemm(stream, s_out, L.v_n, vnv, NN, CS, CS, 0);
    k_vn_lg<<<CDIV(NN * 32, TB), TB, 0, stream>>>(kn, vn_qv, lgv);
    k_vn_softmax<<<BBATCH * 32, 128, 0, stream>>>(lgv, avn);
    k_vn_agg<<<CDIV(BBATCH * NVN * HH * 32, TB), TB, 0, stream>>>(avn, vnv, vn_agg);
    gemm(stream, vn_agg, L.o_vn, vn_tmp, 32, CS, CS, 0);
    ln(stream, vn, vn, vn_tmp, L.ln_vn, 32, CS);
    gemm(stream, s_out, L.q_n, qn, NN, CS, CS, 0);
    gemm(stream, vn, L.k_vn, vn_kv, 32, CS, CS, 0);
    gemm(stream, vn, L.v_vn, vn_vv, 32, CS, CS, 0);
    k_vn_attn2<<<CDIV(NN * HH, TB), TB, 0, stream>>>(qn, vn_kv, vn_vv, agg2);
    gemm(stream, agg2, L.o_n, upd, NN, CS, CS, 0);
    ln(stream, s_out, s_out, upd, L.ln_n, NN, CS);

    // transition
    gemm(stream, s_out, L.nt1, nbuf1, NN, CS, CS, 1);
    gemm(stream, nbuf1, L.nt2, nbuf2, NN, CS, CS, 1);
    gemm(stream, nbuf2, L.nt3, upd, NN, CS, CS, 0);
    ln(stream, s_out, s_out, upd, L.nt_ln, NN, CS);

    // backbone + sidechain
    gemm(stream, s_out, L.bb, u6, NN, CS, 6, 0);
    k_bb_update<<<CDIV(NN, TB), TB, 0, stream>>>(u6, q_out, trans_c);
    gemm(stream, s_out, L.sc, sc_out, NN, CS, CZ, 0, sc_out);

    // edge transition (e_in concat fused into A-load of et1)
    gemm(stream, s_out, L.et_node, ne, NN, CS, 64, 0);
    gemm_edgecat(stream, z_out, ne, e_src, e_dst, L.et1, ebuf1, EE, 2 * CZ, 1);
    gemm(stream, ebuf1, L.et2, z_out, EE, 2 * CZ, CZ, 0);
    ln(stream, z_out, z_out, nullptr, L.et_ln, EE, CZ);
  }

  // ---- epilogue ----
  gemm(stream, s_out, ta1, nbuf1, NN, CS, CS, 1);
  gemm(stream, nbuf1, ta2, nbuf2, NN, CS, 2, 0);
  k_norm_rows<<<CDIV(NN, TB), TB, 0, stream>>>(psi_out, nbuf2, NN, 2);
  k_final_trans<<<CDIV(NN * 3, TB), TB, 0, stream>>>(t_out, trans_c, center);
}
